// TransformerBlock_50646254354971
// MI455X (gfx1250) — matrix operand order repaired and checked by reading, not compiled, not run on hardware
//
#include <hip/hip_runtime.h>

// ---------------- problem constants ----------------
#define B_    2
#define S_    2048
#define D_    1024
#define H_    16
#define DH_   64
#define HKV_  4
#define WIN_  512
#define E_    8
#define KTOP  2
#define FF_   4096
#define T_    (B_ * S_)          // 4096 tokens
#define LMAX  8320               // 8192 assignments + 8*16 padding; /32 = 260 tiles

typedef __bf16 bf16;
typedef __attribute__((ext_vector_type(16))) __bf16 v16bf;
typedef __attribute__((ext_vector_type(8)))  float  v8f;

// ---------------- bf16 helpers ----------------
__device__ __forceinline__ bf16 tobf(float f) {
  unsigned u = __builtin_bit_cast(unsigned, f);
  unsigned short r = (unsigned short)((u + 0x7FFFu + ((u >> 16) & 1u)) >> 16);
  return __builtin_bit_cast(bf16, r);
}
__device__ __forceinline__ float tof(bf16 b) {
  unsigned short s = __builtin_bit_cast(unsigned short, b);
  return __builtin_bit_cast(float, (unsigned)s << 16);
}

// ---------------- async copy helpers (CDNA5 async-to-LDS path) ----------------
// Each lane copies 8 bytes: LDS[lofs] = MEM[src..src+7]; tracked by ASYNCcnt.
__device__ __forceinline__ void async_ld_b64(unsigned lofs, const bf16* src) {
  asm volatile("global_load_async_to_lds_b64 %0, %1, off"
               :: "v"(lofs), "v"(src) : "memory");
}
__device__ __forceinline__ void wait_async1() {
  asm volatile("s_wait_asynccnt 0x1" ::: "memory");
}
__device__ __forceinline__ void wait_async0() {
  asm volatile("s_wait_asynccnt 0x0" ::: "memory");
}

// ---------------- WMMA fragment helpers ----------------
// A-matrix 16x32 bf16 layout: lane m=l&15, hi=l>>4,
// element j -> K = hi*16 + j.  rowptr = &A[row][k0].
__device__ __forceinline__ v16bf load_a16(const bf16* rowptr, int lane) {
  const bf16* p = rowptr + ((lane >> 4) << 4);
  v16bf a;
#pragma unroll
  for (int j = 0; j < 16; ++j) a[j] = p[j];
  return a;
}

// B-matrix 32x16 bf16 from packed layout Bp[(kstep*N + n)*32 + kk],
// lane n=l&15, element j -> kk = (l>>4)*16 + j (contiguous per lane).
__device__ __forceinline__ v16bf load_b_packed(const bf16* bp, int lane) {
  const bf16* p = bp + ((lane & 15) * 32) + ((lane >> 4) << 4);
  v16bf b;
#pragma unroll
  for (int j = 0; j < 16; ++j) b[j] = p[j];
  return b;
}

// B fragment where B[kk][n] = Src[row0+n][col0+kk]  (row-major source used as B^T)
__device__ __forceinline__ v16bf load_b_rows(const bf16* src, int ld, int row0,
                                             int col0, int rowmax, int lane) {
  int r = row0 + (lane & 15);
  if (r > rowmax) r = rowmax;
  const bf16* p = src + (size_t)r * ld + col0 + ((lane >> 4) << 4);
  v16bf b;
#pragma unroll
  for (int j = 0; j < 16; ++j) b[j] = p[j];
  return b;
}

__device__ __forceinline__ v8f wmma_bf16(v16bf a, v16bf b, v8f c) {
  return __builtin_amdgcn_wmma_f32_16x16x32_bf16(false, a, false, b, (short)0, c,
                                                 false, false);
}

// ---------------- weight pack: fp32 row-major [K][N] -> bf16 packed ----------------
__global__ void k_pack(const float* __restrict__ src, bf16* __restrict__ dst,
                       int K, int N) {
  size_t mat = (size_t)K * N;
  const float* s = src + (size_t)blockIdx.z * mat;
  bf16* d = dst + (size_t)blockIdx.z * mat;
  int idx = blockIdx.x * 256 + threadIdx.x;
  if (idx >= K * N) return;
  int k = idx / N, n = idx - k * N;
  d[((size_t)(k >> 5) * N + n) * 32 + (k & 31)] = tobf(s[idx]);
}

// ---------------- RMSNorm: fp32 row -> bf16 normalized row ----------------
__global__ void k_rms(const float* __restrict__ x, const float* __restrict__ w,
                      bf16* __restrict__ out) {
  __shared__ float red[256];
  int t = blockIdx.x, tid = threadIdx.x;
  const float* xr = x + (size_t)t * D_;
  float s = 0.f;
  for (int d = tid; d < D_; d += 256) { float v = xr[d]; s += v * v; }
  red[tid] = s;
  __syncthreads();
  for (int o = 128; o > 0; o >>= 1) {
    if (tid < o) red[tid] += red[tid + o];
    __syncthreads();
  }
  float inv = rsqrtf(red[0] / (float)D_ + 1e-5f);
  bf16* orow = out + (size_t)t * D_;
  for (int d = tid; d < D_; d += 256) orow[d] = tobf(xr[d] * inv * w[d]);
}

// ---------------- generic GEMM: C[M,N] = A_bf16[M,K] * Bp (+res) ----------------
// Block: 8 waves as 2(M)x4(N); block tile 32x256; wave tile 16x64.
// A panel (32x32) staged to LDS via async copy, double buffered.
__global__ void __launch_bounds__(256)
k_gemm(const bf16* __restrict__ A, int lda,
       const bf16* __restrict__ Bp, int N,
       const float* __restrict__ res,
       float* __restrict__ C, int ldc, int K) {
  __shared__ bf16 shA[2][32 * 32];
  int tid = threadIdx.x;
  int lane = tid & 31, w = tid >> 5;
  int mw = w >> 2, nw = w & 3;
  int m0 = blockIdx.y * 32;
  int nb = blockIdx.x * 256 + nw * 64;

  int sr = tid >> 3;                 // staging row 0..31
  int sc = (tid & 7) * 4;            // staging elem 0..28
  const bf16* asrc = A + (size_t)(m0 + sr) * lda + sc;
  unsigned ldst0 = (unsigned)(uintptr_t)&shA[0][sr * 32 + sc];
  unsigned ldst1 = (unsigned)(uintptr_t)&shA[1][sr * 32 + sc];

  async_ld_b64(ldst0, asrc);
  v8f acc[4] = {};
  int nsteps = K >> 5;
  for (int s = 0; s < nsteps; ++s) {
    if (s + 1 < nsteps) {
      async_ld_b64(((s + 1) & 1) ? ldst1 : ldst0, asrc + (s + 1) * 32);
      __builtin_prefetch(Bp + ((size_t)(s + 1) * N + nb) * 32, 0, 1);
      wait_async1();
    } else {
      wait_async0();
    }
    __syncthreads();
    const bf16* ar = &shA[s & 1][(mw * 16 + (lane & 15)) * 32];
    v16bf a = load_a16(ar, lane);
#pragma unroll
    for (int j = 0; j < 4; ++j) {
      v16bf b = load_b_packed(Bp + ((size_t)s * N + nb + j * 16) * 32, lane);
      acc[j] = wmma_bf16(a, b, acc[j]);
    }
    __syncthreads();
  }
  int col0 = nb + (lane & 15);
  int rbase = m0 + mw * 16 + ((lane >> 4) << 3);
#pragma unroll
  for (int j = 0; j < 4; ++j)
#pragma unroll
    for (int i = 0; i < 8; ++i) {
      size_t idx = (size_t)(rbase + i) * ldc + col0 + j * 16;
      float v = acc[j][i];
      if (res) v += res[idx];
      C[idx] = v;
    }
}

// ---------------- RoPE + f32->bf16 (q or k) ----------------
__global__ void k_rope(const float* __restrict__ src, const float* __restrict__ fc,
                       const float* __restrict__ fs, bf16* __restrict__ dst, int nh) {
  int idx = blockIdx.x * 256 + threadIdx.x;   // T * nh * 32 pairs
  int i = idx & 31;
  int rem = idx >> 5;
  int h = rem % nh;
  int t = rem / nh;
  int s = t & (S_ - 1);
  float c = fc[s * 32 + i], sn = fs[s * 32 + i];
  size_t base = (size_t)t * (nh * DH_) + h * DH_ + 2 * i;
  float x1 = src[base], x2 = src[base + 1];
  dst[base]     = tobf(x1 * c - x2 * sn);
  dst[base + 1] = tobf(x1 * sn + x2 * c);
}

// ---------------- V transpose: [T][HKV*64] f32 -> vT[(b*HKV+h)*64 + d][S] bf16 ----------------
__global__ void k_vt(const float* __restrict__ vf, bf16* __restrict__ vT) {
  int idx = blockIdx.x * 256 + threadIdx.x;   // B*HKV*64*S
  int s = idx & (S_ - 1);
  int rem = idx >> 11;
  int d = rem & 63;
  int bh = rem >> 6;
  int b = bh >> 2, h = bh & 3;
  vT[idx] = tobf(vf[((size_t)(b * S_ + s)) * (HKV_ * DH_) + h * DH_ + d]);
}

// ---------------- sliding-window flash attention, one wave per (b,h,16-query tile) ----------------
__global__ void __launch_bounds__(32)
k_attn(const bf16* __restrict__ qb, const bf16* __restrict__ kb,
       const bf16* __restrict__ vT, bf16* __restrict__ ob,
       const int* __restrict__ spp) {
  int lane = threadIdx.x;
  int q0 = blockIdx.x * 16;
  int h = blockIdx.y;
  int b = blockIdx.z;
  int hkv = h >> 2;
  int sp = spp[0];
  int col = lane & 15, hi = lane >> 4;

  const bf16* Arow = qb + (size_t)(b * S_ + q0 + col) * (H_ * DH_) + h * DH_;
  v16bf qa0 = load_a16(Arow + 0, lane);
  v16bf qa1 = load_a16(Arow + 32, lane);
  const bf16* Kb = kb + (size_t)b * S_ * (HKV_ * DH_);
  const bf16* Vb = vT + ((size_t)(b * HKV_ + hkv) * DH_) * S_;

  float mi[8], li[8];
#pragma unroll
  for (int i = 0; i < 8; ++i) { mi[i] = -3.0e38f; li[i] = 0.f; }
  v8f o0 = {}, o1 = {}, o2 = {}, o3 = {};

  __shared__ bf16 pl[16 * 32];

  int qtop = q0 + 15 + sp;
  int chi = qtop < S_ - 1 ? qtop : S_ - 1;
  int e0 = q0 + sp - (WIN_ - 1);
  int clo = e0 > 0 ? (e0 & ~31) : 0;

  for (int c0 = clo; c0 <= chi; c0 += 32) {
    v8f s0 = {}, s1 = {};
    s0 = wmma_bf16(qa0, load_b_rows(Kb, HKV_ * DH_, c0,      hkv * DH_ + 0,  S_ - 1, lane), s0);
    s0 = wmma_bf16(qa1, load_b_rows(Kb, HKV_ * DH_, c0,      hkv * DH_ + 32, S_ - 1, lane), s0);
    s1 = wmma_bf16(qa0, load_b_rows(Kb, HKV_ * DH_, c0 + 16, hkv * DH_ + 0,  S_ - 1, lane), s1);
    s1 = wmma_bf16(qa1, load_b_rows(Kb, HKV_ * DH_, c0 + 16, hkv * DH_ + 32, S_ - 1, lane), s1);

    float p0[8], p1[8], al[8];
#pragma unroll
    for (int i = 0; i < 8; ++i) {
      int r = q0 + hi * 8 + i + sp;
      int kc0 = c0 + col, kc1 = c0 + 16 + col;
      bool a0 = (kc0 <= r) && (kc0 > r - WIN_);
      bool a1 = (kc1 <= r) && (kc1 > r - WIN_);
      float sv0 = a0 ? s0[i] * 0.125f : -3.0e38f;
      float sv1 = a1 ? s1[i] * 0.125f : -3.0e38f;
      float loc = fmaxf(sv0, sv1);
      loc = fmaxf(loc, __shfl_xor(loc, 1));
      loc = fmaxf(loc, __shfl_xor(loc, 2));
      loc = fmaxf(loc, __shfl_xor(loc, 4));
      loc = fmaxf(loc, __shfl_xor(loc, 8));
      float mn = fmaxf(mi[i], loc);
      al[i] = __expf(mi[i] - mn);
      mi[i] = mn;
      p0[i] = a0 ? __expf(sv0 - mn) : 0.f;
      p1[i] = a1 ? __expf(sv1 - mn) : 0.f;
      float rs = p0[i] + p1[i];
      rs += __shfl_xor(rs, 1);
      rs += __shfl_xor(rs, 2);
      rs += __shfl_xor(rs, 4);
      rs += __shfl_xor(rs, 8);
      li[i] = li[i] * al[i] + rs;
    }
#pragma unroll
    for (int i = 0; i < 8; ++i) {
      o0[i] *= al[i]; o1[i] *= al[i]; o2[i] *= al[i]; o3[i] *= al[i];
    }
    // bounce P (16x32) through LDS to re-lay into an A fragment
#pragma unroll
    for (int i = 0; i < 8; ++i) {
      int r = hi * 8 + i;
      pl[r * 32 + col]      = tobf(p0[i]);
      pl[r * 32 + 16 + col] = tobf(p1[i]);
    }
    __syncthreads();
    v16bf pa = load_a16(pl + (size_t)col * 32, lane);
    __syncthreads();
    o0 = wmma_bf16(pa, load_b_rows(Vb, S_, 0,  c0, 63, lane), o0);
    o1 = wmma_bf16(pa, load_b_rows(Vb, S_, 16, c0, 63, lane), o1);
    o2 = wmma_bf16(pa, load_b_rows(Vb, S_, 32, c0, 63, lane), o2);
    o3 = wmma_bf16(pa, load_b_rows(Vb, S_, 48, c0, 63, lane), o3);
  }
#pragma unroll
  for (int i = 0; i < 8; ++i) {
    float inv = 1.f / li[i];
    size_t row = (size_t)(b * S_ + q0 + hi * 8 + i) * (H_ * DH_) + h * DH_;
    ob[row + 0  + col] = tobf(o0[i] * inv);
    ob[row + 16 + col] = tobf(o1[i] * inv);
    ob[row + 32 + col] = tobf(o2[i] * inv);
    ob[row + 48 + col] = tobf(o3[i] * inv);
  }
}

// ---------------- router: softmax + top2 + counts; one wave per token ----------------
__global__ void __launch_bounds__(32)
k_router(const bf16* __restrict__ hf, const float* __restrict__ rw,
         float* __restrict__ probs, int* __restrict__ topi,
         float* __restrict__ gates, int* __restrict__ counts) {
  int t = blockIdx.x, lane = threadIdx.x;
  float acc[E_] = {};
  const bf16* xr = hf + (size_t)t * D_;
  for (int d = lane; d < D_; d += 32) {
    float x = tof(xr[d]);
    const float* r = rw + (size_t)d * E_;
#pragma unroll
    for (int e = 0; e < E_; ++e) acc[e] += x * r[e];
  }
#pragma unroll
  for (int e = 0; e < E_; ++e) {
    float v = acc[e];
    v += __shfl_xor(v, 16); v += __shfl_xor(v, 8); v += __shfl_xor(v, 4);
    v += __shfl_xor(v, 2);  v += __shfl_xor(v, 1);
    acc[e] = v;
  }
  float m = acc[0];
#pragma unroll
  for (int e = 1; e < E_; ++e) m = fmaxf(m, acc[e]);
  float p[E_]; float sum = 0.f;
#pragma unroll
  for (int e = 0; e < E_; ++e) { p[e] = __expf(acc[e] - m); sum += p[e]; }
  float invs = 1.f / sum;
#pragma unroll
  for (int e = 0; e < E_; ++e) p[e] *= invs;
  int i1 = 0;
#pragma unroll
  for (int e = 1; e < E_; ++e) if (p[e] > p[i1]) i1 = e;
  int i2 = (i1 == 0) ? 1 : 0;
#pragma unroll
  for (int e = 0; e < E_; ++e) { if (e != i1 && p[e] > p[i2]) i2 = e; }
  float v1 = p[i1], v2 = p[i2], g = 1.f / (v1 + v2);
  if (lane == 0) {
#pragma unroll
    for (int e = 0; e < E_; ++e) probs[(size_t)t * E_ + e] = p[e];
    topi[t * 2] = i1; topi[t * 2 + 1] = i2;
    gates[t * 2] = v1 * g; gates[t * 2 + 1] = v2 * g;
    atomicAdd(&counts[i1], 1);
    atomicAdd(&counts[i2], 1);
  }
}

__global__ void k_zero(int* counts, int* cursor) {
  int i = threadIdx.x;
  if (i < E_) { counts[i] = 0; cursor[i] = 0; }
}

__global__ void k_scan(const int* counts, int* off) {
  if (threadIdx.x == 0) {
    int a = 0;
    for (int e = 0; e < E_; ++e) { off[e] = a; a += (counts[e] + 15) & ~15; }
    off[E_] = a;
  }
}

__global__ void k_rowmeta(const int* __restrict__ off, int* __restrict__ rowexp,
                          int* __restrict__ list) {
  int r = blockIdx.x * 256 + threadIdx.x;
  if (r >= LMAX) return;
  int e = 0;
  while (e < E_ - 1 && r >= off[e + 1]) ++e;
  rowexp[r] = e;
  list[r] = -1;
}

__global__ void k_assign(const int* __restrict__ topi, const int* __restrict__ off,
                         int* __restrict__ cursor, int* __restrict__ list) {
  int t = blockIdx.x * 256 + threadIdx.x;
  if (t >= T_) return;
  for (int s = 0; s < KTOP; ++s) {
    int e = topi[t * 2 + s];
    int pos = atomicAdd(&cursor[e], 1);
    list[off[e] + pos] = t * 2 + s;
  }
}

// ---------------- MoE up: act = silu(hf*w1) * (hf*w3), grouped rows ----------------
// Block tile 32x256 (8 waves 2x4); gathered A panel async-staged to LDS.
__global__ void __launch_bounds__(256)
k_moe_up(const bf16* __restrict__ hf, const int* __restrict__ list,
         const int* __restrict__ rowexp,
         const bf16* __restrict__ w1p, const bf16* __restrict__ w3p,
         bf16* __restrict__ act) {
  __shared__ bf16 shA[2][32 * 32];
  int tid = threadIdx.x;
  int lane = tid & 31, w = tid >> 5;
  int mw = w >> 2, nw = w & 3;
  int m0 = blockIdx.y * 32;
  int nb = blockIdx.x * 256 + nw * 64;
  int e = rowexp[m0];
  const bf16* B1 = w1p + (size_t)e * D_ * FF_;
  const bf16* B3 = w3p + (size_t)e * D_ * FF_;

  int sr = tid >> 3;
  int sc = (tid & 7) * 4;
  int aidx = list[m0 + sr];
  int tt = aidx >= 0 ? (aidx >> 1) : 0;
  const bf16* asrc = hf + (size_t)tt * D_ + sc;
  unsigned ldst0 = (unsigned)(uintptr_t)&shA[0][sr * 32 + sc];
  unsigned ldst1 = (unsigned)(uintptr_t)&shA[1][sr * 32 + sc];

  async_ld_b64(ldst0, asrc);
  v8f u[4] = {}, g[4] = {};
  const int nsteps = D_ >> 5;
  for (int s = 0; s < nsteps; ++s) {
    if (s + 1 < nsteps) {
      async_ld_b64(((s + 1) & 1) ? ldst1 : ldst0, asrc + (s + 1) * 32);
      __builtin_prefetch(B1 + ((size_t)(s + 1) * FF_ + nb) * 32, 0, 1);
      __builtin_prefetch(B3 + ((size_t)(s + 1) * FF_ + nb) * 32, 0, 1);
      wait_async1();
    } else {
      wait_async0();
    }
    __syncthreads();
    const bf16* ar = &shA[s & 1][(mw * 16 + (lane & 15)) * 32];
    v16bf a = load_a16(ar, lane);
#pragma unroll
    for (int j = 0; j < 4; ++j) {
      size_t boff = ((size_t)s * FF_ + nb + j * 16) * 32;
      u[j] = wmma_bf16(a, load_b_packed(B1 + boff, lane), u[j]);
      g[j] = wmma_bf16(a, load_b_packed(B3 + boff, lane), g[j]);
    }
    __syncthreads();
  }
  int col0 = nb + (lane & 15);
  int rbase = m0 + mw * 16 + ((lane >> 4) << 3);
#pragma unroll
  for (int j = 0; j < 4; ++j)
#pragma unroll
    for (int i = 0; i < 8; ++i) {
      float uu = u[j][i];
      float sv = uu / (1.f + __expf(-uu));
      act[(size_t)(rbase + i) * FF_ + col0 + j * 16] = tobf(sv * g[j][i]);
    }
}

// ---------------- MoE down: ypart[(t,slot)] = gate * act * w2[e] ----------------
__global__ void __launch_bounds__(256)
k_moe_down(const bf16* __restrict__ act, const int* __restrict__ list,
           const int* __restrict__ rowexp,
           const bf16* __restrict__ w2p,
           const float* __restrict__ gates,
           float* __restrict__ ypart) {
  __shared__ bf16 shA[2][32 * 32];
  int tid = threadIdx.x;
  int lane = tid & 31, w = tid >> 5;
  int mw = w >> 2, nw = w & 3;
  int m0 = blockIdx.y * 32;
  int nb = blockIdx.x * 256 + nw * 64;
  int e = rowexp[m0];
  const bf16* B2 = w2p + (size_t)e * FF_ * D_;

  int sr = tid >> 3;
  int sc = (tid & 7) * 4;
  const bf16* asrc = act + (size_t)(m0 + sr) * FF_ + sc;
  unsigned ldst0 = (unsigned)(uintptr_t)&shA[0][sr * 32 + sc];
  unsigned ldst1 = (unsigned)(uintptr_t)&shA[1][sr * 32 + sc];

  async_ld_b64(ldst0, asrc);
  v8f acc[4] = {};
  const int nsteps = FF_ >> 5;
  for (int s = 0; s < nsteps; ++s) {
    if (s + 1 < nsteps) {
      async_ld_b64(((s + 1) & 1) ? ldst1 : ldst0, asrc + (s + 1) * 32);
      __builtin_prefetch(B2 + ((size_t)(s + 1) * D_ + nb) * 32, 0, 1);
      wait_async1();
    } else {
      wait_async0();
    }
    __syncthreads();
    const bf16* ar = &shA[s & 1][(mw * 16 + (lane & 15)) * 32];
    v16bf a = load_a16(ar, lane);
#pragma unroll
    for (int j = 0; j < 4; ++j) {
      v16bf b = load_b_packed(B2 + ((size_t)s * D_ + nb + j * 16) * 32, lane);
      acc[j] = wmma_bf16(a, b, acc[j]);
    }
    __syncthreads();
  }
  int col0 = nb + (lane & 15);
  int rbase = m0 + mw * 16 + ((lane >> 4) << 3);
#pragma unroll
  for (int i = 0; i < 8; ++i) {
    int r = rbase + i;
    int a = list[r];
    if (a >= 0) {
      float gt = gates[a];
#pragma unroll
      for (int j = 0; j < 4; ++j)
        ypart[(size_t)a * D_ + col0 + j * 16] = acc[j][i] * gt;
    }
  }
}

// ---------------- final: out = h + ypart[t,0] + ypart[t,1] ----------------
__global__ void k_final(const float* __restrict__ h, const float* __restrict__ ypart,
                        float* __restrict__ out) {
  int idx = blockIdx.x * 256 + threadIdx.x;   // T_*D_
  int t = idx >> 10, d = idx & 1023;
  out[idx] = h[idx] + ypart[(size_t)(t * 2) * D_ + d] + ypart[(size_t)(t * 2 + 1) * D_ + d];
}

// ---------------- aux loss ----------------
__global__ void k_probsum(const float* __restrict__ probs, float* __restrict__ psum) {
  __shared__ float red[256];
  int e = blockIdx.x, tid = threadIdx.x;
  float s = 0.f;
  for (int t = tid; t < T_; t += 256) s += probs[(size_t)t * E_ + e];
  red[tid] = s;
  __syncthreads();
  for (int o = 128; o > 0; o >>= 1) {
    if (tid < o) red[tid] += red[tid + o];
    __syncthreads();
  }
  if (tid == 0) psum[e] = red[0];
}

__global__ void k_aux(const int* __restrict__ counts, const float* __restrict__ psum,
                      float* __restrict__ out_aux) {
  if (threadIdx.x == 0) {
    float a = 0.f;
    for (int e = 0; e < E_; ++e)
      a += ((float)counts[e] / (float)(T_ * KTOP)) * (psum[e] / (float)T_);
    out_aux[0] = (float)E_ * a;
  }
}

// ---------------- host launcher ----------------
extern "C" void kernel_launch(void* const* d_in, const int* in_sizes, int n_in,
                              void* d_out, int out_size, void* d_ws, size_t ws_size,
                              hipStream_t stream) {
  (void)in_sizes; (void)n_in; (void)out_size; (void)ws_size;
  const float* x   = (const float*)d_in[0];
  const float* fc  = (const float*)d_in[1];
  const float* fs  = (const float*)d_in[2];
  const float* anw = (const float*)d_in[3];
  const float* fnw = (const float*)d_in[4];
  const float* wq  = (const float*)d_in[5];
  const float* wk  = (const float*)d_in[6];
  const float* wv  = (const float*)d_in[7];
  const float* wo  = (const float*)d_in[8];
  const float* rw  = (const float*)d_in[9];
  const float* w1  = (const float*)d_in[10];
  const float* w3  = (const float*)d_in[11];
  const float* w2  = (const float*)d_in[12];
  const int*   sp  = (const int*)d_in[13];

  char* p = (char*)d_ws;
  auto alloc = [&](size_t bytes) -> void* {
    void* r = (void*)p;
    p += (bytes + 255) & ~(size_t)255;
    return r;
  };

  bf16* wqp  = (bf16*)alloc((size_t)D_ * D_ * 2);
  bf16* wkp  = (bf16*)alloc((size_t)D_ * (HKV_ * DH_) * 2);
  bf16* wvp  = (bf16*)alloc((size_t)D_ * (HKV_ * DH_) * 2);
  bf16* wop  = (bf16*)alloc((size_t)D_ * D_ * 2);
  bf16* w1p  = (bf16*)alloc((size_t)E_ * D_ * FF_ * 2);
  bf16* w3p  = (bf16*)alloc((size_t)E_ * D_ * FF_ * 2);
  bf16* w2p  = (bf16*)alloc((size_t)E_ * FF_ * D_ * 2);
  bf16* hn   = (bf16*)alloc((size_t)T_ * D_ * 2);
  float* qf  = (float*)alloc((size_t)T_ * (H_ * DH_) * 4);
  float* kf  = (float*)alloc((size_t)T_ * (HKV_ * DH_) * 4);
  float* vf  = (float*)alloc((size_t)T_ * (HKV_ * DH_) * 4);
  bf16* qbb  = (bf16*)alloc((size_t)T_ * (H_ * DH_) * 2);
  bf16* kbb  = (bf16*)alloc((size_t)T_ * (HKV_ * DH_) * 2);
  bf16* vTb  = (bf16*)alloc((size_t)B_ * HKV_ * DH_ * S_ * 2 + 256);
  bf16* obb  = (bf16*)alloc((size_t)T_ * (H_ * DH_) * 2);
  float* hbuf = (float*)alloc((size_t)T_ * D_ * 4);
  bf16* hfb  = (bf16*)alloc((size_t)T_ * D_ * 2);
  float* probs = (float*)alloc((size_t)T_ * E_ * 4);
  int* topi  = (int*)alloc((size_t)T_ * KTOP * 4);
  float* gates = (float*)alloc((size_t)T_ * KTOP * 4);
  int* counts = (int*)alloc(E_ * 4);
  int* cursor = (int*)alloc(E_ * 4);
  int* off    = (int*)alloc((E_ + 1) * 4);
  int* rowexp = (int*)alloc(LMAX * 4);
  int* list   = (int*)alloc(LMAX * 4);
  bf16* act   = (bf16*)alloc((size_t)LMAX * FF_ * 2);
  float* ypart = (float*)alloc((size_t)T_ * KTOP * D_ * 4);
  float* psum  = (float*)alloc(E_ * 4);

  float* out = (float*)d_out;
  float* out_aux = out + (size_t)T_ * D_;

  k_zero<<<1, 32, 0, stream>>>(counts, cursor);

  // pack weights to bf16 WMMA-friendly layout
  k_pack<<<dim3((D_ * D_) / 256, 1, 1), 256, 0, stream>>>(wq, wqp, D_, D_);
  k_pack<<<dim3((D_ * HKV_ * DH_) / 256, 1, 1), 256, 0, stream>>>(wk, wkp, D_, HKV_ * DH_);
  k_pack<<<dim3((D_ * HKV_ * DH_) / 256, 1, 1), 256, 0, stream>>>(wv, wvp, D_, HKV_ * DH_);
  k_pack<<<dim3((D_ * D_) / 256, 1, 1), 256, 0, stream>>>(wo, wop, D_, D_);
  k_pack<<<dim3((D_ * FF_) / 256, 1, E_), 256, 0, stream>>>(w1, w1p, D_, FF_);
  k_pack<<<dim3((D_ * FF_) / 256, 1, E_), 256, 0, stream>>>(w3, w3p, D_, FF_);
  k_pack<<<dim3((FF_ * D_) / 256, 1, E_), 256, 0, stream>>>(w2, w2p, FF_, D_);

  // attention path
  k_rms<<<T_, 256, 0, stream>>>(x, anw, hn);
  k_gemm<<<dim3((H_ * DH_) / 256, T_ / 32), 256, 0, stream>>>(hn, D_, wqp, H_ * DH_, nullptr, qf, H_ * DH_, D_);
  k_gemm<<<dim3(1, T_ / 32), 256, 0, stream>>>(hn, D_, wkp, HKV_ * DH_, nullptr, kf, HKV_ * DH_, D_);
  k_gemm<<<dim3(1, T_ / 32), 256, 0, stream>>>(hn, D_, wvp, HKV_ * DH_, nullptr, vf, HKV_ * DH_, D_);
  k_rope<<<(T_ * H_ * 32) / 256, 256, 0, stream>>>(qf, fc, fs, qbb, H_);
  k_rope<<<(T_ * HKV_ * 32) / 256, 256, 0, stream>>>(kf, fc, fs, kbb, HKV_);
  k_vt<<<(B_ * HKV_ * DH_ * S_) / 256, 256, 0, stream>>>(vf, vTb);
  k_attn<<<dim3(S_ / 16, H_, B_), 32, 0, stream>>>(qbb, kbb, vTb, obb, sp);
  k_gemm<<<dim3(D_ / 256, T_ / 32), 256, 0, stream>>>(obb, H_ * DH_, wop, D_, x, hbuf, D_, H_ * DH_);

  // MoE path
  k_rms<<<T_, 256, 0, stream>>>(hbuf, fnw, hfb);
  k_router<<<T_, 32, 0, stream>>>(hfb, rw, probs, topi, gates, counts);
  k_scan<<<1, 1, 0, stream>>>(counts, off);
  k_rowmeta<<<(LMAX + 255) / 256, 256, 0, stream>>>(off, rowexp, list);
  k_assign<<<T_ / 256, 256, 0, stream>>>(topi, off, cursor, list);
  k_moe_up<<<dim3(FF_ / 256, LMAX / 32), 256, 0, stream>>>(hfb, list, rowexp, w1p, w3p, act);
  k_moe_down<<<dim3(D_ / 256, LMAX / 32), 256, 0, stream>>>(act, list, rowexp, w2p, gates, ypart);
  k_final<<<(T_ * D_) / 256, 256, 0, stream>>>(hbuf, ypart, out);

  // aux loss
  k_probsum<<<E_, 256, 0, stream>>>(probs, psum);
  k_aux<<<1, 1, 0, stream>>>(counts, psum, out_aux);
}
